// PGNN_49563922596334
// MI455X (gfx1250) — matrix operand order repaired and checked by reading, not compiled, not run on hardware
//
#include <hip/hip_runtime.h>
#include <math.h>

typedef __attribute__((ext_vector_type(16))) _Float16 v16h;
typedef __attribute__((ext_vector_type(8)))  _Float16 v8h;
typedef __attribute__((ext_vector_type(4)))  _Float16 v4h;
typedef __attribute__((ext_vector_type(8)))  float    v8f;
typedef __attribute__((ext_vector_type(4)))  float    fvec4;

#define BN_EPS 1e-5f

// =====================================================================
// WMMA GEMM: C[M,N] = act( (A*scale+shift)[M,K] @ W[K,N] + bias ) (+resid)
// act: 0=none, 1=relu, 2=clip[0,1].  relu applied BEFORE residual (PGN order).
// Block = 256 threads = 8 waves (4M x 2N), wave tile 64x16 -> macrotile 256x32.
// Per K-step a wave does 4 WMMAs sharing one B fragment; all fragment loads
// are issued before the WMMA chain so LDS latency overlaps XDL execution.
// A staged [256][40] f16 (80B row stride -> all v8h slices 16B aligned),
// B staged transposed [n][k].  Double-buffered LDS, one barrier per K-step.
// Pure-copy f16 A staging uses global_load_async_to_lds_b128 (ASYNCcnt).
// =====================================================================
template <typename TA, typename TC>
__global__ __launch_bounds__(256)
void wmma_gemm_kernel(const TA* __restrict__ A,
                      const float* __restrict__ scale,
                      const float* __restrict__ shift,
                      const float* __restrict__ W,
                      const float* __restrict__ bias,
                      const float* __restrict__ resid,
                      TC* __restrict__ C,
                      int M, int K, int N, int act)
{
    __shared__ __attribute__((aligned(16))) _Float16 As[2][256][40];
    __shared__ __attribute__((aligned(16))) _Float16 Bs[2][32][40]; // [n][k]

    const int tid  = threadIdx.x;
    const int lane = tid & 31;
    const int wave = tid >> 5;
    const int wm   = wave >> 1;           // 0..3 -> 64-row band
    const int wn   = wave & 1;            // 0..1 -> 16-col band
    const int mBlock = blockIdx.x * 256;
    const int nBlock = blockIdx.y * 32;

    const int bcol = tid & 31;            // n within B tile
    const int brow = (tid >> 5) * 4;      // k within B tile

    auto stage = [&](int k0, int buf) {
        // ---- A tile 256x32: 4 (row,group) pairs per thread;
        //      4 consecutive lanes cover one row's 64B (coalesced).
#pragma unroll
        for (int p = 0; p < 4; ++p) {
            const int  r    = (tid >> 2) + p * 64;   // 0..255
            const int  g2   = tid & 3;               // 8-half group
            const int  kb   = k0 + g2 * 8;
            const int  m_ok = (mBlock + r) < M;
            const long rowBase = (long)(mBlock + r) * K;
            if constexpr (sizeof(TA) == 2) {
                if (m_ok && !scale && (kb + 8 <= K)) {
                    // pure f16 copy: async DMA global -> LDS, 16B per lane
                    unsigned lds = (unsigned)(size_t)&As[buf][r][g2 * 8];
                    unsigned long long ga =
                        (unsigned long long)(const void*)(A + rowBase + kb);
                    asm volatile("global_load_async_to_lds_b128 %0, %1, off"
                                 :: "v"(lds), "v"(ga) : "memory");
                    continue;
                }
            }
            v8h out;
            if (m_ok && (kb + 8 <= K)) {
                float va[8];
                if constexpr (sizeof(TA) == 2) {
                    v8h t = *(const v8h*)(A + rowBase + kb);
#pragma unroll
                    for (int j = 0; j < 8; ++j) va[j] = (float)t[j];
                } else {
                    fvec4 a0 = *(const fvec4*)(A + rowBase + kb);
                    fvec4 a1 = *(const fvec4*)(A + rowBase + kb + 4);
#pragma unroll
                    for (int j = 0; j < 4; ++j) { va[j] = a0[j]; va[4 + j] = a1[j]; }
                }
                if (scale) {
#pragma unroll
                    for (int j = 0; j < 8; ++j) va[j] = va[j] * scale[kb + j] + shift[kb + j];
                }
#pragma unroll
                for (int j = 0; j < 8; ++j) out[j] = (_Float16)va[j];
            } else {
#pragma unroll
                for (int j = 0; j < 8; ++j) {
                    int k = kb + j;
                    float v = 0.f;
                    if (m_ok && k < K) {
                        v = (float)A[rowBase + k];
                        if (scale) v = v * scale[k] + shift[k];
                    }
                    out[j] = (_Float16)v;
                }
            }
            *(v8h*)&As[buf][r][g2 * 8] = out;
        }
        // ---- B tile 32x32, stored transposed [n][k]
        {
            const int n = nBlock + bcol;
            v4h ob;
#pragma unroll
            for (int j = 0; j < 4; ++j) {
                int k = k0 + brow + j;
                float v = (k < K && n < N) ? W[(long)k * N + n] : 0.f;
                ob[j] = (_Float16)v;
            }
            *(v4h*)&Bs[buf][bcol][brow] = ob;
        }
    };

    v8f acc[4];
#pragma unroll
    for (int s = 0; s < 4; ++s) acc[s] = (v8f){0.f,0.f,0.f,0.f,0.f,0.f,0.f,0.f};

    const int nk = (K + 31) / 32;
    stage(0, 0);

    const int rowA0 = wm * 64 + (lane & 15);
    const int kbA   = (lane >= 16) ? 8 : 0;
    const int colB  = wn * 16 + (lane & 15);
    const int kbB   = (lane >= 16) ? 8 : 0;

    for (int kt = 0; kt < nk; ++kt) {
        asm volatile("s_wait_asynccnt 0x0" ::: "memory"); // publish async A stage
        __syncthreads();
        if (kt + 1 < nk) stage((kt + 1) * 32, (kt + 1) & 1);
        const int buf = kt & 1;
        // issue ALL fragment loads first (distinct registers), then WMMA chain
        v16h bf2, afr[4];
        *((v8h*)&bf2)     = *(const v8h*)&Bs[buf][colB][kbB];
        *((v8h*)&bf2 + 1) = *(const v8h*)&Bs[buf][colB][16 + kbB];
#pragma unroll
        for (int s = 0; s < 4; ++s) {
            const int r = rowA0 + s * 16;
            *((v8h*)&afr[s])     = *(const v8h*)&As[buf][r][kbA];
            *((v8h*)&afr[s] + 1) = *(const v8h*)&As[buf][r][16 + kbA];
        }
#pragma unroll
        for (int s = 0; s < 4; ++s) {
            acc[s] = __builtin_amdgcn_wmma_f32_16x16x32_f16(
                         false, afr[s], false, bf2, (short)0, acc[s], false, false);
        }
    }

    // C layout: element i -> row = mbase+i, col = lane&15 (mbase=8 for lanes>=16)
    const int n = nBlock + wn * 16 + (lane & 15);
    if (n < N) {
#pragma unroll
        for (int s = 0; s < 4; ++s) {
            const int mrow = mBlock + wm * 64 + s * 16 + ((lane >= 16) ? 8 : 0);
#pragma unroll
            for (int i = 0; i < 8; ++i) {
                int m = mrow + i;
                if (m < M) {
                    float v = acc[s][i] + bias[n];
                    if (act == 1) v = fmaxf(v, 0.f);
                    if (resid) v += resid[(long)m * N + n];
                    if (act == 2) v = fminf(fmaxf(v, 0.f), 1.f);
                    C[(long)m * N + n] = (TC)v;
                }
            }
        }
    }
}

// =====================================================================
// Gather / concat kernels
// =====================================================================
__global__ void concat_edge_kernel(const float* __restrict__ ef, int We,
                                   const float* __restrict__ nf, int Wn,
                                   const float* __restrict__ gf, int Wg,
                                   const int* __restrict__ src,
                                   const int* __restrict__ dst,
                                   const int* __restrict__ egid,
                                   float* __restrict__ out, int rows, int rowOff)
{
    int Ktot = We + 2 * Wn + Wg;
    long total = (long)rows * Ktot;
    for (long idx = blockIdx.x * (long)blockDim.x + threadIdx.x; idx < total;
         idx += (long)gridDim.x * blockDim.x) {
        int r = (int)(idx / Ktot), c = (int)(idx % Ktot);
        int ge = r + rowOff;
        float v;
        if (c < We)              v = ef[(long)ge * We + c];
        else if (c < We + Wn)    v = nf[(long)src[ge] * Wn + (c - We)];
        else if (c < We + 2*Wn)  v = nf[(long)dst[ge] * Wn + (c - We - Wn)];
        else                     v = gf[(long)egid[ge] * Wg + (c - We - 2*Wn)];
        out[idx] = v;
    }
}

__global__ void concat_node_kernel(const float* __restrict__ nfp, int Wn,
                                   const float* __restrict__ agg,
                                   const float* __restrict__ gfp, int Wg,
                                   const int* __restrict__ ngid,
                                   float* __restrict__ out, int N)
{
    int Ktot = Wn + 32 + Wg;
    long total = (long)N * Ktot;
    for (long idx = blockIdx.x * (long)blockDim.x + threadIdx.x; idx < total;
         idx += (long)gridDim.x * blockDim.x) {
        int r = (int)(idx / Ktot), c = (int)(idx % Ktot);
        float v;
        if (c < Wn)           v = nfp[(long)r * Wn + c];
        else if (c < Wn + 32) v = agg[(long)r * 32 + (c - Wn)];
        else                  v = gfp[(long)ngid[r] * Wg + (c - Wn - 32)];
        out[idx] = v;
    }
}

__global__ void concat_global_kernel(const float* __restrict__ g, int Wg,
                                     const float* __restrict__ me,
                                     const float* __restrict__ mn,
                                     float* __restrict__ out, int G)
{
    int Ktot = Wg + 64;
    int total = G * Ktot;
    int idx = blockIdx.x * blockDim.x + threadIdx.x;
    if (idx < total) {
        int r = idx / Ktot, c = idx % Ktot;
        float v;
        if (c < Wg)           v = g[r * Wg + c];
        else if (c < Wg + 32) v = me[r * 32 + (c - Wg)];
        else                  v = mn[r * 32 + (c - Wg - 32)];
        out[idx] = v;
    }
}

// =====================================================================
// BatchNorm statistics (LDS partials -> one global atomic per block/feature)
// =====================================================================
__global__ void bn_stats_kernel(const float* __restrict__ X, long M, int K,
                                float* __restrict__ sums, float* __restrict__ sqs)
{
    __shared__ float s0[128], s1[128];
    for (int k = threadIdx.x; k < K; k += blockDim.x) { s0[k] = 0.f; s1[k] = 0.f; }
    __syncthreads();
    long total = M * K;
    for (long idx = blockIdx.x * (long)blockDim.x + threadIdx.x; idx < total;
         idx += (long)gridDim.x * blockDim.x) {
        float v = X[idx];
        int k = (int)(idx % K);
        atomicAdd(&s0[k], v);
        atomicAdd(&s1[k], v * v);
    }
    __syncthreads();
    for (int k = threadIdx.x; k < K; k += blockDim.x) {
        atomicAdd(&sums[k], s0[k]);
        atomicAdd(&sqs[k],  s1[k]);
    }
}

__global__ void bn_finalize_kernel(const float* __restrict__ sums,
                                   const float* __restrict__ sqs, float invM,
                                   const float* __restrict__ gamma,
                                   const float* __restrict__ beta,
                                   float* __restrict__ scale,
                                   float* __restrict__ shift, int K)
{
    int k = blockIdx.x * blockDim.x + threadIdx.x;
    if (k < K) {
        float m   = sums[k] * invM;
        float var = fmaxf(sqs[k] * invM - m * m, 0.f);
        float sc  = gamma[k] * rsqrtf(var + BN_EPS);
        scale[k] = sc;
        shift[k] = beta[k] - m * sc;
    }
}

// =====================================================================
// Segment ops
// =====================================================================
__global__ void seg_sum_kernel(const float* __restrict__ x, long M, int F,
                               const int* __restrict__ ids,
                               float* __restrict__ out, float* __restrict__ cnt)
{
    long total = M * F;
    for (long idx = blockIdx.x * (long)blockDim.x + threadIdx.x; idx < total;
         idx += (long)gridDim.x * blockDim.x) {
        int r = (int)(idx / F), f = (int)(idx % F);
        int s = ids[r];
        atomicAdd(&out[(long)s * F + f], x[idx]);
        if (cnt && f == 0) atomicAdd(&cnt[s], 1.f);
    }
}

__global__ void seg_div_kernel(float* __restrict__ out, const float* __restrict__ cnt,
                               int S, int F, float minv)
{
    long total = (long)S * F;
    for (long idx = blockIdx.x * (long)blockDim.x + threadIdx.x; idx < total;
         idx += (long)gridDim.x * blockDim.x) {
        out[idx] /= fmaxf(cnt[idx / F], minv);
    }
}

__global__ void fill_kernel(float* __restrict__ p, float v, int n)
{
    int i = blockIdx.x * blockDim.x + threadIdx.x;
    if (i < n) p[i] = v;
}

// =====================================================================
// Attention (last PGN block)
// =====================================================================
__global__ void att_logit_kernel(const float* __restrict__ e,
                                 const float* __restrict__ w,
                                 const float* __restrict__ b,
                                 float* __restrict__ logit, int E)
{
    int i = blockIdx.x * blockDim.x + threadIdx.x;
    if (i < E) {
        const float* row = e + (long)i * 32;
        float s = b[0];
#pragma unroll
        for (int k = 0; k < 32; ++k) s += row[k] * w[k];
        logit[i] = s;
    }
}

__device__ inline void atomicMaxF(float* addr, float val)
{
    unsigned* u = (unsigned*)addr;
    unsigned old = *u;
    while (__uint_as_float(old) < val) {
        unsigned assumed = old;
        old = atomicCAS(u, assumed, __float_as_uint(val));
        if (old == assumed) break;
    }
}

__global__ void seg_max_kernel(const float* __restrict__ logit,
                               const int* __restrict__ dst,
                               float* __restrict__ smax, int E)
{
    int i = blockIdx.x * blockDim.x + threadIdx.x;
    if (i < E) atomicMaxF(&smax[dst[i]], logit[i]);
}

__global__ void att_weight_kernel(const float* __restrict__ logit,
                                  const float* __restrict__ smax,
                                  const int* __restrict__ dst,
                                  float* __restrict__ wexp,
                                  float* __restrict__ denom, int E)
{
    int i = blockIdx.x * blockDim.x + threadIdx.x;
    if (i < E) {
        float w = __expf(logit[i] - smax[dst[i]]);
        wexp[i] = w;
        atomicAdd(&denom[dst[i]], w);
    }
}

__global__ void att_agg_kernel(const float* __restrict__ wexp,
                               const float* __restrict__ e_new,
                               const int* __restrict__ dst,
                               float* __restrict__ agg, int E)
{
    long total = (long)E * 32;
    for (long idx = blockIdx.x * (long)blockDim.x + threadIdx.x; idx < total;
         idx += (long)gridDim.x * blockDim.x) {
        int r = (int)(idx / 32), f = (int)(idx % 32);
        atomicAdd(&agg[(long)dst[r] * 32 + f], wexp[r] * e_new[idx]);
    }
}

// =====================================================================
// Host orchestration
// =====================================================================
struct MLPP { const float *W0,*W1,*W2,*b0,*b1,*b2,*gamma,*beta; };

extern "C" void kernel_launch(void* const* d_in, const int* in_sizes, int n_in,
                              void* d_out, int out_size, void* d_ws, size_t ws_size,
                              hipStream_t stream)
{
    const int*   src  = (const int*)  d_in[0];
    const int*   dst  = (const int*)  d_in[1];
    const int*   ngid = (const int*)  d_in[2];
    const int*   egid = (const int*)  d_in[3];
    const float* nf   = (const float*)d_in[4];
    const float* ef   = (const float*)d_in[5];
    const float* u    = (const float*)d_in[6];

    const int E = in_sizes[0];
    const int N = in_sizes[4] / 8;   // NIN = 8
    const int G = in_sizes[6] / 4;   // GIN = 4

    // ---- unpack params (jax tree-flatten order: sorted dict keys, list order)
    int pi = 7;
    auto take = [&]() { return (const float*)d_in[pi++]; };
    auto takeMLP = [&](bool norm) {
        MLPP m{};
        m.W0 = take(); m.W1 = take(); m.W2 = take();
        if (norm) m.beta = take();
        m.b0 = take(); m.b1 = take(); m.b2 = take();
        if (norm) m.gamma = take();
        return m;
    };
    MLPP em[4], gm[4], nm[4];
    const float *att_w = nullptr, *att_b = nullptr;
    for (int i = 0; i < 4; ++i) {
        if (i == 3) { att_b = take(); att_w = take(); }
        em[i] = takeMLP(true);
        gm[i] = takeMLP(true);
        nm[i] = takeMLP(true);
    }
    MLPP reg = takeMLP(false);

    // ---- workspace carve-out
    const int CH  = 65536;
    const long CHR = (N > CH) ? (long)N : CH;
    char* ws = (char*)d_ws;
    size_t off = 0;
    auto alloc = [&](size_t bytes) -> void* {
        void* p = (void*)(ws + off);
        off += (bytes + 255) & ~(size_t)255;
        return p;
    };
    float* ebuf[2] = { (float*)alloc((size_t)E*32*4), (float*)alloc((size_t)E*32*4) };
    float* nbuf[2] = { (float*)alloc((size_t)N*32*4), (float*)alloc((size_t)N*32*4) };
    float* gbuf[2] = { (float*)alloc((size_t)G*32*4), (float*)alloc((size_t)G*32*4) };
    float*     agg   = (float*)alloc((size_t)N*32*4);
    float*     X     = (float*)alloc((size_t)CHR*128*4);
    _Float16*  h1    = (_Float16*)alloc((size_t)CHR*256*2);   // f16 hidden acts
    _Float16*  h2    = (_Float16*)alloc((size_t)CHR*128*2);
    float* sums  = (float*)alloc(128*4);
    float* sqs   = (float*)alloc(128*4);
    float* scale = (float*)alloc(128*4);
    float* shift = (float*)alloc(128*4);
    float* meanE = (float*)alloc((size_t)G*32*4);
    float* meanN = (float*)alloc((size_t)G*32*4);
    float* cntG  = (float*)alloc((size_t)G*4);
    float* cntN  = (float*)alloc((size_t)N*4);
    float* logit = (float*)alloc((size_t)E*4);
    float* wexp  = (float*)alloc((size_t)E*4);
    float* denom = (float*)alloc((size_t)N*4);
    float* smax  = (float*)alloc((size_t)N*4);
    (void)ws_size; (void)n_in; (void)out_size;

    auto cdiv = [](long a, long b) { return (int)((a + b - 1) / b); };
    auto gemmFH = [&](const float* A, const float* sc, const float* sh,
                      const float* Wp, const float* bp, _Float16* C,
                      int M, int K, int Nc) {
        dim3 g(cdiv(M, 256), cdiv(Nc, 32));
        wmma_gemm_kernel<float, _Float16><<<g, 256, 0, stream>>>(
            A, sc, sh, Wp, bp, nullptr, C, M, K, Nc, 1);
    };
    auto gemmHH = [&](const _Float16* A, const float* Wp, const float* bp,
                      _Float16* C, int M, int K, int Nc) {
        dim3 g(cdiv(M, 256), cdiv(Nc, 32));
        wmma_gemm_kernel<_Float16, _Float16><<<g, 256, 0, stream>>>(
            A, nullptr, nullptr, Wp, bp, nullptr, C, M, K, Nc, 1);
    };
    auto gemmHF = [&](const _Float16* A, const float* Wp, const float* bp,
                      const float* res, float* C, int M, int K, int Nc, int act) {
        dim3 g(cdiv(M, 256), cdiv(Nc, 32));
        wmma_gemm_kernel<_Float16, float><<<g, 256, 0, stream>>>(
            A, nullptr, nullptr, Wp, bp, res, C, M, K, Nc, act);
    };
    auto statsBlocks = [&](long total) { int b = cdiv(total, 256); return b > 512 ? 512 : b; };

    const float* ce = ef; const float* cn = nf; const float* cg = u;
    int We = 8, Wn = 8, Wg = 4;

    for (int i = 0; i < 4; ++i) {
        float* ne = ebuf[i & 1];
        float* nn = nbuf[i & 1];
        float* ng = gbuf[i & 1];
        const int Ke = We + 2 * Wn + Wg;
        const int Kn = Wn + 32 + Wg;
        const int Kg = Wg + 64;
        const int nch = (E + CH - 1) / CH;

        // ---------- edge MLP: BN stats over all edges (streamed in chunks)
        hipMemsetAsync(sums, 0, 128*4, stream);
        hipMemsetAsync(sqs,  0, 128*4, stream);
        for (int c = 0; c < nch; ++c) {
            int rows = (c == nch-1) ? (E - c*CH) : CH;
            long tot = (long)rows * Ke;
            concat_edge_kernel<<<cdiv(tot,256), 256, 0, stream>>>(
                ce, We, cn, Wn, cg, Wg, src, dst, egid, X, rows, c*CH);
            bn_stats_kernel<<<statsBlocks(tot), 256, 0, stream>>>(X, rows, Ke, sums, sqs);
        }
        bn_finalize_kernel<<<1, 128, 0, stream>>>(sums, sqs, 1.f/(float)E,
                                                  em[i].gamma, em[i].beta, scale, shift, Ke);
        // ---------- edge MLP compute (BN folded into first GEMM's A-load)
        for (int c = 0; c < nch; ++c) {
            int rows = (c == nch-1) ? (E - c*CH) : CH;
            long tot = (long)rows * Ke;
            concat_edge_kernel<<<cdiv(tot,256), 256, 0, stream>>>(
                ce, We, cn, Wn, cg, Wg, src, dst, egid, X, rows, c*CH);
            gemmFH(X, scale, shift, em[i].W0, em[i].b0, h1, rows, Ke, 256);
            gemmHH(h1, em[i].W1, em[i].b1, h2, rows, 256, 128);
            const float* res = (i >= 1) ? ce + (long)c*CH*32 : nullptr;
            gemmHF(h2, em[i].W2, em[i].b2, res, ne + (long)c*CH*32, rows, 128, 32, 1);
        }

        // ---------- edge -> node aggregation
        if (i == 3) {
            att_logit_kernel<<<cdiv(E,256), 256, 0, stream>>>(ne, att_w, att_b, logit, E);
            fill_kernel<<<cdiv(N,256), 256, 0, stream>>>(smax, -3.4e38f, N);
            seg_max_kernel<<<cdiv(E,256), 256, 0, stream>>>(logit, dst, smax, E);
            hipMemsetAsync(denom, 0, (size_t)N*4, stream);
            att_weight_kernel<<<cdiv(E,256), 256, 0, stream>>>(logit, smax, dst, wexp, denom, E);
            hipMemsetAsync(agg, 0, (size_t)N*32*4, stream);
            att_agg_kernel<<<cdiv((long)E*32,256), 256, 0, stream>>>(wexp, ne, dst, agg, E);
            seg_div_kernel<<<cdiv((long)N*32,256), 256, 0, stream>>>(agg, denom, N, 32, BN_EPS);
        } else {
            hipMemsetAsync(agg,  0, (size_t)N*32*4, stream);
            hipMemsetAsync(cntN, 0, (size_t)N*4,   stream);
            seg_sum_kernel<<<cdiv((long)E*32,256), 256, 0, stream>>>(ne, E, 32, dst, agg, cntN);
            seg_div_kernel<<<cdiv((long)N*32,256), 256, 0, stream>>>(agg, cntN, N, 32, 1.f);
        }

        // ---------- node MLP
        {
            long tot = (long)N * Kn;
            concat_node_kernel<<<cdiv(tot,256), 256, 0, stream>>>(cn, Wn, agg, cg, Wg, ngid, X, N);
            hipMemsetAsync(sums, 0, 128*4, stream);
            hipMemsetAsync(sqs,  0, 128*4, stream);
            bn_stats_kernel<<<statsBlocks(tot), 256, 0, stream>>>(X, N, Kn, sums, sqs);
            bn_finalize_kernel<<<1, 128, 0, stream>>>(sums, sqs, 1.f/(float)N,
                                                      nm[i].gamma, nm[i].beta, scale, shift, Kn);
            gemmFH(X, scale, shift, nm[i].W0, nm[i].b0, h1, N, Kn, 256);
            gemmHH(h1, nm[i].W1, nm[i].b1, h2, N, 256, 128);
            gemmHF(h2, nm[i].W2, nm[i].b2, (i >= 1) ? cn : nullptr, nn, N, 128, 32, 1);
        }

        // ---------- global MLP
        {
            hipMemsetAsync(meanE, 0, (size_t)G*32*4, stream);
            hipMemsetAsync(cntG,  0, (size_t)G*4,   stream);
            seg_sum_kernel<<<cdiv((long)E*32,256), 256, 0, stream>>>(ne, E, 32, egid, meanE, cntG);
            seg_div_kernel<<<cdiv((long)G*32,256), 256, 0, stream>>>(meanE, cntG, G, 32, 1.f);
            hipMemsetAsync(meanN, 0, (size_t)G*32*4, stream);
            hipMemsetAsync(cntG,  0, (size_t)G*4,   stream);
            seg_sum_kernel<<<cdiv((long)N*32,256), 256, 0, stream>>>(nn, N, 32, ngid, meanN, cntG);
            seg_div_kernel<<<cdiv((long)G*32,256), 256, 0, stream>>>(meanN, cntG, G, 32, 1.f);

            concat_global_kernel<<<cdiv((long)G*Kg,256), 256, 0, stream>>>(cg, Wg, meanE, meanN, X, G);
            hipMemsetAsync(sums, 0, 128*4, stream);
            hipMemsetAsync(sqs,  0, 128*4, stream);
            bn_stats_kernel<<<statsBlocks((long)G*Kg), 256, 0, stream>>>(X, G, Kg, sums, sqs);
            bn_finalize_kernel<<<1, 128, 0, stream>>>(sums, sqs, 1.f/(float)G,
                                                      gm[i].gamma, gm[i].beta, scale, shift, Kg);
            gemmFH(X, scale, shift, gm[i].W0, gm[i].b0, h1, G, Kg, 256);
            gemmHH(h1, gm[i].W1, gm[i].b1, h2, G, 256, 128);
            gemmHF(h2, gm[i].W2, gm[i].b2, (i >= 1) ? cg : nullptr, ng, G, 128, 32, 1);
        }

        ce = ne; cn = nn; cg = ng;
        We = Wn = Wg = 32;
    }

    // ---------- readout MLP: 32 -> 256 -> 128 -> 1, clip[0,1]
    gemmFH(cn, nullptr, nullptr, reg.W0, reg.b0, h1, N, 32, 256);
    gemmHH(h1, reg.W1, reg.b1, h2, N, 256, 128);
    gemmHF(h2, reg.W2, reg.b2, nullptr, (float*)d_out, N, 128, 1, 2);
}
